// ReformerAttention_47304769798174
// MI455X (gfx1250) — matrix operand order repaired and checked by reading, not compiled, not run on hardware
//
#include <hip/hip_runtime.h>
#include <math.h>

typedef __attribute__((ext_vector_type(16))) _Float16 v16h;
typedef __attribute__((ext_vector_type(8)))  float    v8f;

#define B_   2
#define T_   4096
#define H_   8
#define E_   64
#define BH_  16
#define NB_  64
#define NH_  8
#define BS_  64
#define SMEM_BYTES 91136

// ---------------------------------------------------------------------------
// Kernel 1: LSH hashing. One thread per (hash, bh, t).
// ---------------------------------------------------------------------------
__global__ __launch_bounds__(256) void lsh_hash_kernel(
    const float* __restrict__ qk, const float* __restrict__ rot,
    int* __restrict__ buckets, float* __restrict__ invnorm) {
  int gid  = blockIdx.x * 256 + threadIdx.x;          // hash-major
  int hash = gid / (BH_ * T_);
  int rem  = gid - hash * (BH_ * T_);
  int bh   = rem / T_;
  int t    = rem - bh * T_;
  int b = bh / H_, hh = bh - b * H_;
  const float* qp = qk + (((size_t)b * T_ + t) * H_ + hh) * E_;
  float acc[32];
#pragma unroll
  for (int i = 0; i < 32; ++i) acc[i] = 0.f;
  float ss = 0.f;
  for (int e = 0; e < E_; ++e) {
    float q = qp[e];
    ss += q * q;
    const float* rp = rot + ((size_t)e * NH_ + hash) * 32;
#pragma unroll
    for (int i = 0; i < 32; ++i) acc[i] = fmaf(q, rp[i], acc[i]);
  }
  float bv = acc[0]; int bi = 0;
#pragma unroll
  for (int i = 1; i < 32; ++i) if (acc[i] > bv) { bv = acc[i]; bi = i; }
#pragma unroll
  for (int i = 0; i < 32; ++i) { float nv = -acc[i]; if (nv > bv) { bv = nv; bi = 32 + i; } }
  buckets[gid] = bi;
  if (hash == 0) invnorm[bh * T_ + t] = rsqrtf(ss);
}

// ---------------------------------------------------------------------------
// Kernel 2: stable counting sort per (hash, bh) sequence of 4096 bucket ids.
// ---------------------------------------------------------------------------
__global__ __launch_bounds__(256) void lsh_sort_kernel(
    const int* __restrict__ buckets, int* __restrict__ perm) {
  __shared__ int lb[T_];
  __shared__ int hist[16 * 64];
  __shared__ int startb[64];
  int tid = threadIdx.x;
  int seq = blockIdx.x;
  const int* src = buckets + (size_t)seq * T_;
  for (int i = tid; i < T_; i += 256) lb[i] = src[i];
  for (int i = tid; i < 1024; i += 256) hist[i] = 0;
  __syncthreads();
  int base = tid * 16;
  int chunk = tid >> 4;
  for (int k = 0; k < 16; ++k) atomicAdd(&hist[chunk * 64 + lb[base + k]], 1);
  __syncthreads();
  if (tid < 64) { int s = 0; for (int c = 0; c < 16; ++c) s += hist[c * 64 + tid]; startb[tid] = s; }
  __syncthreads();
  if (tid == 0) { int run = 0; for (int bk = 0; bk < 64; ++bk) { int x = startb[bk]; startb[bk] = run; run += x; } }
  __syncthreads();
  if (tid < 64) { int run = startb[tid]; for (int c = 0; c < 16; ++c) { int x = hist[c * 64 + tid]; hist[c * 64 + tid] = run; run += x; } }
  __syncthreads();
  int cb = chunk << 8;
  int* dst = perm + (size_t)seq * T_;
  for (int k = 0; k < 16; ++k) {
    int t = base + k;
    int bb = lb[t];
    int cnt = 0;
    for (int j = cb; j < t; ++j) cnt += (lb[j] == bb);   // stable tie-break by index
    dst[hist[chunk * 64 + bb] + cnt] = t;
  }
}

// ---------------------------------------------------------------------------
// Fragment helpers: all fragments are contiguous, 32B-aligned v16h in LDS.
// A-tiles are stored fragment-major: [tile][lane][16 halves].
// Contiguous-k A layout used here for element (m, k) of a 16x32 tile:
//   lh = (k>>4)&1;  hh = k&15;  lane = m + 16*lh
// B-fragments read 16 consecutive K-halves for column n at lane n (+16 for K+16).
// ---------------------------------------------------------------------------
__device__ inline v16h ldsv16(const _Float16* p) { return *(const v16h*)p; }

// ---------------------------------------------------------------------------
// Kernel 3: bucketed attention for one hash round. One block per (bh, bucket).
// ---------------------------------------------------------------------------
__global__ __launch_bounds__(256) void lsh_attn_kernel(
    const float* __restrict__ qk, const float* __restrict__ vin,
    const int* __restrict__ perm, const float* __restrict__ invnorm,
    float* __restrict__ o_slice, float* __restrict__ lse_slice, int hash) {
  extern __shared__ char smem[];
  _Float16* QA = (_Float16*)smem;              // 8 A-tiles  * 512 halves =  8 KB
  _Float16* Ks = (_Float16*)(smem + 8192);     // 128x64 row-major f16    = 16 KB
  _Float16* Vt = (_Float16*)(smem + 24576);    // 64x128 V-transposed f16 = 16 KB
  float*    Ss = (float*)   (smem + 40960);    // 64x128 f32 scores       = 32 KB
  float*    Os = (float*)   (smem + 40960);    // reused: 64x64 f32 output (16 KB)
  _Float16* PA = (_Float16*)(smem + 73728);    // 16 A-tiles * 512 halves = 16 KB
  int*      qidx = (int*)   (smem + 90112);    // 64
  int*      kidx = (int*)   (smem + 90368);    // 128

  int tid = threadIdx.x;
  int bh  = blockIdx.x / NB_;
  int n   = blockIdx.x - bh * NB_;
  int b = bh / H_, hh = bh - b * H_;
  const int* pbase = perm + ((size_t)hash * BH_ + bh) * T_;

  if (tid < 64) {
    qidx[tid] = pbase[n * BS_ + tid];
  } else if (tid < 192) {
    int j = tid - 64;
    int src = (j < 64) ? (n * BS_ + j)
                       : (((n + NB_ - 1) % NB_) * BS_ + (j - 64)); // rolled prev bucket
    kidx[j] = pbase[src];
  }
  __syncthreads();

  // Gather Q into fragment-major A-tile layout (f16).
  for (int i = tid; i < 64 * 64; i += 256) {
    int r = i >> 6, e = i & 63;
    int t = qidx[r];
    float qv = qk[(((size_t)b * T_ + t) * H_ + hh) * E_ + e];
    int mtile = r >> 4, m = r & 15;
    int kc = e >> 5, k = e & 31;
    int lh = (k >> 4) & 1;
    int hq = k & 15;
    int lane = m + (lh << 4);
    QA[(((mtile << 1) | kc) * 32 + lane) * 16 + hq] = (_Float16)qv;
  }
  // Gather K (normalized, row-major) and V (transposed) into LDS.
  for (int i = tid; i < 128 * 64; i += 256) {
    int r = i >> 6, e = i & 63;
    int t = kidx[r];
    float sc = invnorm[bh * T_ + t];
    size_t gi = (((size_t)b * T_ + t) * H_ + hh) * E_ + e;
    Ks[r * 64 + e]  = (_Float16)(qk[gi] * sc);
    Vt[e * 128 + r] = (_Float16)vin[gi];
  }
  __syncthreads();

  int wave = tid >> 5, lane = tid & 31;
  int lh = (lane >> 4) & 1, ln = lane & 15;

  // ---- S = Q * K^T : 4x8 tiles of 16x16, 8 waves x 4 tiles each -------------
  {
    int mtile = wave >> 1;
    int nbase = (wave & 1) * 4;
    for (int nt = 0; nt < 4; ++nt) {
      int ntile = nbase + nt;
      v8f acc = {};
#pragma unroll
      for (int kc = 0; kc < 2; ++kc) {
        v16h a  = ldsv16(QA + ((mtile * 2 + kc) * 32 + lane) * 16);
        // B(k,n) = K[ntile*16+n][kc*32 + lh*16 + q] : contiguous 16 halves
        v16h bf = ldsv16(Ks + (ntile * 16 + ln) * 64 + kc * 32 + lh * 16);
        acc = __builtin_amdgcn_wmma_f32_16x16x32_f16(false, a, false, bf,
                                                     (short)0, acc, false, false);
      }
#pragma unroll
      for (int r = 0; r < 8; ++r) {
        int gm = mtile * 16 + r + lh * 8;
        int gn = ntile * 16 + ln;
        float val = acc[r] * 0.125f;                 // softmax_temp = 1/sqrt(64)
        if (qidx[gm] == kidx[gn]) val = -50000.0f;   // self-attn mask
        Ss[gm * 128 + gn] = val;
      }
    }
  }
  __syncthreads();

  // ---- row softmax + lse; write P in fragment-major A-tile layout ----------
  if (tid < 64) {
    const float* row = Ss + tid * 128;
    float m = row[0];
    for (int j = 1; j < 128; ++j) m = fmaxf(m, row[j]);
    float s = 0.f;
    for (int j = 0; j < 128; ++j) s += __expf(row[j] - m);
    float inv = 1.0f / s;
    int mtile = tid >> 4, mm = tid & 15;
    for (int j = 0; j < 128; ++j) {
      float p = __expf(row[j] - m) * inv;
      int kc = j >> 5, k = j & 31;
      int plh = (k >> 4) & 1;
      int hq = k & 15;
      int plane = mm + (plh << 4);
      PA[((mtile * 4 + kc) * 32 + plane) * 16 + hq] = (_Float16)p;
    }
    lse_slice[(size_t)bh * T_ + qidx[tid]] = m + __logf(s);
  }
  __syncthreads();

  // ---- O = P * V : 4x4 tiles, 8 waves x 2 tiles each -----------------------
  v8f oacc[2];
#pragma unroll
  for (int tt = 0; tt < 2; ++tt) {
    int ti = wave + tt * 8;
    int mtile = ti >> 2, ntile = ti & 3;
    v8f acc = {};
#pragma unroll
    for (int kc = 0; kc < 4; ++kc) {
      v16h a  = ldsv16(PA + ((mtile * 4 + kc) * 32 + lane) * 16);
      // B(k,n) = V[kc*32 + lh*16 + q][ntile*16+n] = Vt[(ntile*16+n)*128 + ...]
      v16h bf = ldsv16(Vt + (ntile * 16 + ln) * 128 + kc * 32 + lh * 16);
      acc = __builtin_amdgcn_wmma_f32_16x16x32_f16(false, a, false, bf,
                                                   (short)0, acc, false, false);
    }
    oacc[tt] = acc;
  }
  __syncthreads();   // S buffer no longer needed; reuse as f32 O staging
#pragma unroll
  for (int tt = 0; tt < 2; ++tt) {
    int ti = wave + tt * 8;
    int mtile = ti >> 2, ntile = ti & 3;
#pragma unroll
    for (int r = 0; r < 8; ++r) {
      Os[(mtile * 16 + r + lh * 8) * 64 + ntile * 16 + ln] = oacc[tt][r];
    }
  }
  __syncthreads();
  // Coalesced scatter: each 64-thread group writes one token's 64 floats.
  for (int i = tid; i < 64 * 64; i += 256) {
    int r = i >> 6, e = i & 63;
    o_slice[((size_t)bh * T_ + qidx[r]) * E_ + e] = Os[i];
  }
}

// ---------------------------------------------------------------------------
// Kernel 4: accumulate num += e^{lse}*o, den += e^{lse} (init overwrites).
// ---------------------------------------------------------------------------
__global__ __launch_bounds__(256) void lsh_accum_kernel(
    const float* __restrict__ o_slice, const float* __restrict__ lse_slice,
    float* __restrict__ num, float* __restrict__ den, int init) {
  int idx = blockIdx.x * 256 + threadIdx.x;   // BH*T*E
  int bt = idx >> 6;
  int e  = idx & 63;
  float w = __expf(lse_slice[bt]);
  float contrib = w * o_slice[idx];
  if (init) num[idx] = contrib; else num[idx] += contrib;
  if (e == 0) { if (init) den[bt] = w; else den[bt] += w; }
}

// ---------------------------------------------------------------------------
// Kernel 5: out[b,t,h,e] = num[bh,t,e] / den[bh,t]
// ---------------------------------------------------------------------------
__global__ __launch_bounds__(256) void lsh_final_kernel(
    const float* __restrict__ num, const float* __restrict__ den,
    float* __restrict__ out) {
  int idx = blockIdx.x * 256 + threadIdx.x;   // B*T*H*E
  int e = idx & 63;
  int h = (idx >> 6) & 7;
  int t = (idx >> 9) & 4095;
  int b = idx >> 21;
  int bt = (b * H_ + h) * T_ + t;
  out[idx] = num[(size_t)bt * E_ + e] / den[bt];
}

extern "C" void kernel_launch(void* const* d_in, const int* in_sizes, int n_in,
                              void* d_out, int out_size, void* d_ws, size_t ws_size,
                              hipStream_t stream) {
  const float* qk  = (const float*)d_in[0];
  const float* vin = (const float*)d_in[1];
  const float* rot = (const float*)d_in[2];
  float* out = (float*)d_out;

  char* ws = (char*)d_ws;
  size_t off = 0;
  auto carve = [&](size_t bytes) -> char* {
    char* p = ws + off;
    off += (bytes + 255) & ~(size_t)255;
    return p;
  };
  int*   buckets   = (int*)  carve((size_t)NH_ * BH_ * T_ * 4);        // 2 MB
  int*   perm      = (int*)  carve((size_t)NH_ * BH_ * T_ * 4);        // 2 MB
  float* invnorm   = (float*)carve((size_t)BH_ * T_ * 4);              // 256 KB
  float* o_slice   = (float*)carve((size_t)BH_ * T_ * E_ * 4);         // 16 MB
  float* lse_slice = (float*)carve((size_t)BH_ * T_ * 4);              // 256 KB
  float* num       = (float*)carve((size_t)BH_ * T_ * E_ * 4);         // 16 MB
  float* den       = (float*)carve((size_t)BH_ * T_ * 4);              // 256 KB
  if (off > ws_size) return;  // workspace too small; nothing safe to do

  lsh_hash_kernel<<<(NH_ * BH_ * T_) / 256, 256, 0, stream>>>(qk, rot, buckets, invnorm);
  lsh_sort_kernel<<<NH_ * BH_, 256, 0, stream>>>(buckets, perm);
  for (int h = 0; h < NH_; ++h) {
    lsh_attn_kernel<<<BH_ * NB_, 256, SMEM_BYTES, stream>>>(
        qk, vin, perm, invnorm, o_slice, lse_slice, h);
    lsh_accum_kernel<<<(BH_ * T_ * E_) / 256, 256, 0, stream>>>(
        o_slice, lse_slice, num, den, h == 0);
  }
  lsh_final_kernel<<<(B_ * T_ * H_ * E_) / 256, 256, 0, stream>>>(num, den, out);
}
